// CrossAttention_63084479644164
// MI455X (gfx1250) — matrix operand order repaired and checked by reading, not compiled, not run on hardware
//
#include <hip/hip_runtime.h>

// Problem constants (match reference)
#define Bq  256
#define Ssz 64
#define Esz 256
#define Hn  8
#define dsz 32
#define Dd  2048   // Ssz * dsz

typedef __attribute__((ext_vector_type(16))) __bf16 v16bf;
typedef __attribute__((ext_vector_type(8)))  float  v8f;
typedef __attribute__((ext_vector_type(4)))  int    v4i;

typedef __attribute__((address_space(1))) v4i glob_v4i;   // global
typedef __attribute__((address_space(3))) v4i lds_v4i;    // LDS

union FragU { uint4 u[2]; v16bf v; };                 // 32B WMMA A/B fragment
union CvtU  { __bf16 h[16]; v16bf v; };               // f32->bf16 staging

// --- CDNA5 async global->LDS copy (ASYNCcnt path), guarded for portability ---
__device__ __forceinline__ void async_copy_b128(const void* gsrc, void* ldst) {
#if __has_builtin(__builtin_amdgcn_global_load_async_to_lds_b128)
  __builtin_amdgcn_global_load_async_to_lds_b128(
      (glob_v4i*)gsrc, (lds_v4i*)ldst, 0, 0);
#else
  *(uint4*)ldst = *(const uint4*)gsrc;   // sync fallback through VGPRs
#endif
}

__device__ __forceinline__ void wait_async_lds() {
#if __has_builtin(__builtin_amdgcn_s_wait_asynccnt)
  __builtin_amdgcn_s_wait_asynccnt(0);
#elif __has_builtin(__builtin_amdgcn_global_load_async_to_lds_b128)
  asm volatile("s_wait_asynccnt 0x0" ::: "memory");
#endif
}

// ---------------------------------------------------------------------------
// Kernel 1: split heads + fp32 -> bf16 pack:  [B,S,E] -> [H,B,D] bf16
// ---------------------------------------------------------------------------
__global__ __launch_bounds__(256) void pack_heads_bf16(
    const float* __restrict__ dec, const float* __restrict__ enc,
    __bf16* __restrict__ xd, __bf16* __restrict__ xe) {
  int i   = blockIdx.x * 256 + threadIdx.x;     // over H*B*D = 2^22
  int h   = i >> 19;                            // B*D = 2^19
  int rem = i & ((1 << 19) - 1);
  int b   = rem >> 11;                          // D = 2^11
  int ii  = rem & (Dd - 1);
  int s   = ii >> 5, j = ii & 31;
  int src = ((b * Ssz + s) * Esz) + h * dsz + j;
  xd[i] = (__bf16)dec[src];
  xe[i] = (__bf16)enc[src];
}

// ---------------------------------------------------------------------------
// Kernel 2: per-head GEMM  Y[h] = Xbf16[h] (256xD) * W[h]^T (DxD) + bias[h]
// Both tiles async-copied global->LDS (double buffered); W stays fp32 in LDS
// and is converted to bf16 at fragment build; v_wmma_f32_16x16x32_bf16 with
// fp32 accumulation. Block tile 128x128, K-step 32, 8 waves, 64x32 C / wave.
// ---------------------------------------------------------------------------
#define TM  128
#define TN  128
#define TK  32
#define LDA 40   // bf16 elems per A-tile LDS row (80B, 16B aligned)
#define LDW 36   // f32  elems per W-tile LDS row (144B, 16B aligned)

__global__ __launch_bounds__(256) void qkv_gemm_bf16(
    const __bf16* __restrict__ Abf,   // [H][B][D] bf16
    const float* __restrict__ W,      // [H][D][D] fp32 (row = out dim)
    const float* __restrict__ bias,   // [H][D]
    float* __restrict__ Y) {          // [H][B][D] fp32
  __shared__ __bf16 Alds[2][TM * LDA];   // 2 x 10,240 B
  __shared__ float  Wlds[2][TN * LDW];   // 2 x 18,432 B

  const int h    = blockIdx.z;
  const int nblk = blockIdx.x;
  const int mblk = blockIdx.y;
  const int tid  = threadIdx.x;
  const int lane = tid & 31;
  const int wave = tid >> 5;
  const int wm   = wave & 1;     // 2 wave-rows
  const int wn   = wave >> 1;    // 4 wave-cols
  const int rA   = lane & 15;
  const int hi16 = lane >> 4;

  const __bf16* A  = Abf + ((size_t)h * Bq + (size_t)mblk * TM) * Dd;
  const float*  Wh = W + (size_t)h * Dd * Dd + (size_t)nblk * TN * Dd;

  v8f acc[4][2];
#pragma unroll
  for (int m = 0; m < 4; ++m)
#pragma unroll
    for (int n = 0; n < 2; ++n)
#pragma unroll
      for (int e = 0; e < 8; ++e) acc[m][n][e] = 0.0f;

  // issue one K-tile (A: 8KB bf16, W: 16KB f32) as async b128 copies
  auto issue_tile = [&](int k0, int buf) {
#pragma unroll
    for (int c = 0; c < 2; ++c) {          // A: 512 chunks of 8 bf16
      int cid = c * 256 + tid;
      int row = cid >> 2, chk = cid & 3;
      async_copy_b128(A + (size_t)row * Dd + k0 + chk * 8,
                      &Alds[buf][row * LDA + chk * 8]);
    }
#pragma unroll
    for (int c = 0; c < 4; ++c) {          // W: 1024 chunks of 4 f32
      int cid = c * 256 + tid;
      int row = cid >> 3, chk = cid & 7;
      async_copy_b128(Wh + (size_t)row * Dd + k0 + chk * 4,
                      &Wlds[buf][row * LDW + chk * 4]);
    }
  };

  const int NK = Dd / TK;   // 64
  issue_tile(0, 0);

  for (int i = 0; i < NK; ++i) {
    const int cur = i & 1;
    wait_async_lds();          // this wave's async writes to buf[cur] done
    __syncthreads();           // all waves' writes visible; prev reads retired

    if (i + 1 < NK) {
      issue_tile((i + 1) * TK, cur ^ 1);
      if (i + 2 < NK)          // global_prefetch_b8 two tiles ahead
        __builtin_prefetch(Wh + (size_t)(tid >> 1) * Dd + (size_t)(i + 2) * TK, 0, 0);
    }

    // --- fragments: A per documented 16-bit 16x32 striping; W cvt->bf16 ---
    const __bf16* Ab = &Alds[cur][0];
    const float*  Wb = &Wlds[cur][0];
    FragU afr[4];
#pragma unroll
    for (int m = 0; m < 4; ++m) {
      const __bf16* p = Ab + (wm * 64 + m * 16 + rA) * LDA + hi16 * 8;
      afr[m].u[0] = *(const uint4*)(p);        // K = kb .. kb+7
      afr[m].u[1] = *(const uint4*)(p + 16);   // K = kb+16 .. kb+23
    }
    v16bf bfr[2];
#pragma unroll
    for (int n = 0; n < 2; ++n) {
      const float* p = Wb + (wn * 32 + n * 16 + rA) * LDW + hi16 * 8;
      CvtU cv;
#pragma unroll
      for (int e = 0; e < 16; ++e) cv.h[e] = (__bf16)p[e < 8 ? e : e + 8];   // v_cvt_pk_bf16_f32
      bfr[n] = cv.v;
    }

#pragma unroll
    for (int m = 0; m < 4; ++m)
#pragma unroll
      for (int n = 0; n < 2; ++n)
        acc[m][n] = __builtin_amdgcn_wmma_f32_16x16x32_bf16(
            false, afr[m].v, false, bfr[n], (short)0, acc[m][n], false, false);

    __syncthreads();           // buf[cur] fully consumed before it is refilled
  }

  // --- epilogue: C layout VGPR r -> M = r + 8*(lane>=16), N = lane&15 ---
  float* Yh = Y + (size_t)h * Bq * Dd;
#pragma unroll
  for (int m = 0; m < 4; ++m) {
    int row0 = mblk * TM + wm * 64 + m * 16 + hi16 * 8;
#pragma unroll
    for (int n = 0; n < 2; ++n) {
      int col  = nblk * TN + wn * 32 + n * 16 + rA;
      float bv = bias[(size_t)h * Dd + col];
#pragma unroll
      for (int r = 0; r < 8; ++r)
        Yh[(size_t)(row0 + r) * Dd + col] = acc[m][n][r] + bv;
    }
  }
}

// ---------------------------------------------------------------------------
// Kernel 3: attention per (h,b). scores = Q K^T; softmax over the QUERY axis
// (reference axis=2); Z = attn @ V; scale 1/sqrt(d) applied AFTER attn@V.
// ---------------------------------------------------------------------------
__global__ __launch_bounds__(256) void attn_softmax_q(
    const float* __restrict__ QKV,   // [3][H][B][D]
    float* __restrict__ out) {       // [B][S][E]
  __shared__ float Qs[Ssz][dsz + 1];
  __shared__ float Ks[Ssz][dsz + 1];
  __shared__ float Vs[Ssz][dsz + 1];
  __shared__ float Sc[Ssz][Ssz + 1];

  const int hb  = blockIdx.x;
  const int h   = hb >> 8;         // B = 256
  const int b   = hb & 255;
  const int tid = threadIdx.x;
  const size_t HBD = (size_t)Hn * Bq * Dd;
  const float* Qp = QKV + ((size_t)h * Bq + b) * Dd;
  const float* Kp = Qp + HBD;
  const float* Vp = Qp + 2 * HBD;

#pragma unroll
  for (int c = 0; c < 8; ++c) {
    int idx = c * 256 + tid;
    int s = idx >> 5, j = idx & 31;
    Qs[s][j] = Qp[idx];
    Ks[s][j] = Kp[idx];
    Vs[s][j] = Vp[idx];
  }
  __syncthreads();

#pragma unroll
  for (int c = 0; c < 16; ++c) {
    int e = c * 256 + tid;
    int q = e >> 6, k = e & 63;
    float acc = 0.0f;
#pragma unroll
    for (int j = 0; j < dsz; ++j) acc += Qs[q][j] * Ks[k][j];
    Sc[q][k] = acc;
  }
  __syncthreads();

  // softmax over q for each key column k (reference: softmax over query axis)
  if (tid < Ssz) {
    int k = tid;
    float mx = -3.402823466e+38f;
    for (int q = 0; q < Ssz; ++q) mx = fmaxf(mx, Sc[q][k]);
    float sum = 0.0f;
    for (int q = 0; q < Ssz; ++q) {
      float e = __expf(Sc[q][k] - mx);
      Sc[q][k] = e;
      sum += e;
    }
    float inv = 1.0f / sum;
    for (int q = 0; q < Ssz; ++q) Sc[q][k] *= inv;
  }
  __syncthreads();

  const float scale = 0.17677669529663687f;   // 1/sqrt(32)
#pragma unroll
  for (int c = 0; c < 8; ++c) {
    int e = c * 256 + tid;
    int q = e >> 5, j = e & 31;
    float z = 0.0f;
#pragma unroll
    for (int k = 0; k < Ssz; ++k) z += Sc[q][k] * Vs[k][j];
    out[((size_t)b * Ssz + q) * Esz + h * dsz + j] = z * scale;
  }
}

// ---------------------------------------------------------------------------
extern "C" void kernel_launch(void* const* d_in, const int* in_sizes, int n_in,
                              void* d_out, int out_size, void* d_ws, size_t ws_size,
                              hipStream_t stream) {
  (void)in_sizes; (void)n_in; (void)out_size; (void)ws_size;
  const float* dec = (const float*)d_in[0];
  const float* enc = (const float*)d_in[1];
  const float* WQ  = (const float*)d_in[2];
  const float* bQ  = (const float*)d_in[3];
  const float* WK  = (const float*)d_in[4];
  const float* bK  = (const float*)d_in[5];
  const float* WV  = (const float*)d_in[6];
  const float* bV  = (const float*)d_in[7];
  float* out = (float*)d_out;

  const size_t HBD = (size_t)Hn * Bq * Dd;          // 4,194,304 elements
  char* ws = (char*)d_ws;
  __bf16* xd = (__bf16*)ws;                          // 8.39 MB
  __bf16* xe = (__bf16*)(ws + HBD * 2);              // 8.39 MB
  float* QKV = (float*)(ws + 2 * (HBD * 2));         // 50.3 MB

  pack_heads_bf16<<<(unsigned)(HBD / 256), 256, 0, stream>>>(dec, enc, xd, xe);

  dim3 g(Dd / TN, Bq / TM, Hn);   // (16, 2, 8)
  qkv_gemm_bf16<<<g, 256, 0, stream>>>(xd, WQ, bQ, QKV);
  qkv_gemm_bf16<<<g, 256, 0, stream>>>(xe, WK, bK, QKV + HBD);
  qkv_gemm_bf16<<<g, 256, 0, stream>>>(xe, WV, bV, QKV + 2 * HBD);

  attn_softmax_q<<<Hn * Bq, 256, 0, stream>>>(QKV, out);
}
